// S5Layer_53034256171309
// MI455X (gfx1250) — matrix operand order repaired and checked by reading, not compiled, not run on hardware
//
#include <hip/hip_runtime.h>
#include <hip/hip_bf16.h>

// ---------------------------------------------------------------------------
// S5 layer on MI455X (gfx1250), wave32.
//   prep (Lbar, dt-scaled/concat weights, bf16 hi/lo splits)
//   -> GEMM1 (BU = x @ Wb^T)          [split-bf16 WMMA, f32 acc]
//   -> scan  (complex recurrence, 512 chains, batched prefetch)
//   -> GEMM2 (Y = S @ Wc^T + D*x)     [writes Y hi/lo]
//   -> GEMM3 (OUT = Y @ Wo^T + b)
// GEMM: B (weight) tiles double-buffered in LDS via async global->LDS copies
// (ASYNCcnt), A streamed from global; v_wmma_f32_16x16x32_bf16 everywhere.
// ---------------------------------------------------------------------------

typedef __bf16 bf16;
typedef __attribute__((ext_vector_type(16))) __bf16 bf16x16;
typedef __attribute__((ext_vector_type(8)))  __bf16 bf16x8;
typedef __attribute__((ext_vector_type(8)))  float  v8f;
typedef __attribute__((ext_vector_type(4)))  int    i32x4;

#define AS1 __attribute__((address_space(1)))
#define AS3 __attribute__((address_space(3)))
typedef AS1 i32x4* gvec_ptr;   // global-AS int4*  (async builtin src)
typedef AS3 i32x4* lvec_ptr;   // lds-AS int4*     (async builtin dst)

#if __has_builtin(__builtin_amdgcn_global_load_async_to_lds_b128)
#define HAVE_ASYNC_LDS 1
#else
#define HAVE_ASYNC_LDS 0
#endif

#define D_MODEL 1024
#define D_STATE 64
#define BATCH   8
#define SEQLEN  2048
#define M_ROWS  (BATCH * SEQLEN)          // 16384
#define OUT_ELEMS ((size_t)M_ROWS * D_MODEL)

#define KSLAB   64                         // K per LDS stage (2 WMMA k-steps)
#define BPITCH  72                         // LDS row pitch (bank-conflict pad)

// ---- workspace layout (bytes) ----------------------------------------------
static const size_t O_LBAR = 0;                          // 128 floats
static const size_t O_WBH  = 1024;                       // 128x1024 bf16
static const size_t SZ_WB  = (size_t)128 * 1024 * 2;
static const size_t O_WBL  = O_WBH + SZ_WB;
static const size_t O_WCH  = O_WBL + SZ_WB;              // 1024x128 bf16
static const size_t SZ_WC  = (size_t)1024 * 128 * 2;
static const size_t O_WCL  = O_WCH + SZ_WC;
static const size_t O_WOH  = O_WCL + SZ_WC;              // 1024x1024 bf16
static const size_t SZ_WO  = (size_t)1024 * 1024 * 2;
static const size_t O_WOL  = O_WOH + SZ_WO;
static const size_t O_BU   = O_WOL + SZ_WO;              // 16384x128 f32
static const size_t SZ_BU  = (size_t)M_ROWS * 128 * 4;
static const size_t O_SH   = O_BU + SZ_BU;               // 16384x128 bf16
static const size_t SZ_S   = (size_t)M_ROWS * 128 * 2;
static const size_t O_SL   = O_SH + SZ_S;
static const size_t O_XYH  = O_SL + SZ_S;                // x_hi, later Y_hi
static const size_t SZ_XY  = (size_t)M_ROWS * 1024 * 2;
static const size_t O_XYL  = O_XYH + SZ_XY;              // x_lo, later Y_lo

// ---- helpers ---------------------------------------------------------------
__device__ inline bf16 f2bf(float f) { return (bf16)f; }

union Frag16 { bf16x16 v; bf16x8 h[2]; };

// A-matrix 16x32 bf16 fragment (row-major A, leading dim = K):
// lane holds K = kb..kb+7 (v0..3) and kb+16..kb+23 (v4..7), kb = k0 + 8*(lane>=16)
__device__ inline bf16x16 load_a_frag(const bf16* __restrict__ base, int ldk,
                                      int row, int k0, int lane) {
  const bf16* p = base + (size_t)row * ldk + k0 + ((lane >> 4) << 3);
  Frag16 f;
  f.h[0] = *(const bf16x8*)(p);
  f.h[1] = *(const bf16x8*)(p + 16);
  return f.v;
}

// B-matrix 32x16 bf16 fragment from LDS tile (row = n-local, pitch BPITCH):
// lane: n = lane%16 (+tile offset), K = kb..kb+7 and kb+16..kb+23, kb = kk + 8*(lane/16).
__device__ inline bf16x16 load_b_frag_lds(const bf16* tile, int nloc, int kk, int lane) {
  const bf16* p = tile + (nloc + (lane & 15)) * BPITCH + kk + ((lane >> 4) << 3);
  Frag16 f;
  f.h[0] = *(const bf16x8*)(p);
  f.h[1] = *(const bf16x8*)(p + 16);
  return f.v;
}

__device__ inline void wait_async0() {
#if HAVE_ASYNC_LDS
#if __has_builtin(__builtin_amdgcn_s_wait_asynccnt)
  __builtin_amdgcn_s_wait_asynccnt(0);
#else
  asm volatile("s_wait_asynccnt 0x0" ::: "memory");
#endif
#endif
}

// Stage one 64 x KSLAB bf16 tile (8KB) of row-major G (ld=K) into LDS.
// 512 chunks of 16B; 256 threads -> 2 chunks each.
__device__ inline void stage_tile(const bf16* __restrict__ G, bf16* S,
                                  int nblk0, int K, int kslab, int tid) {
#pragma unroll
  for (int i = 0; i < 2; ++i) {
    int c   = tid + (i << 8);          // 0..511
    int row = c >> 3;
    int off = (c & 7) << 3;            // element offset (8 bf16 = 16B)
    const bf16* g = G + (size_t)(nblk0 + row) * K + kslab + off;
    bf16* l = S + row * BPITCH + off;
#if HAVE_ASYNC_LDS
    // signature (per clang diagnostic): (int4 AS1* src, lds dst, Ii offset, Ii cpol)
    __builtin_amdgcn_global_load_async_to_lds_b128(
        (gvec_ptr)(i32x4*)g, (lvec_ptr)(i32x4*)l, 0, 0);
#else
    *(bf16x8*)l = *(const bf16x8*)g;   // sync fallback: global->reg->ds
#endif
  }
}

// ---- prep kernels ----------------------------------------------------------
__global__ void k_lambda(const float* __restrict__ log_Lam_real,
                         const float* __restrict__ Lam_imag,
                         const float* __restrict__ log_dt,
                         float* __restrict__ Lbar) {
  int n = threadIdx.x;
  if (n < D_STATE) {
    float dt   = __expf(log_dt[0]);
    float lamr = -__expf(log_Lam_real[n]);
    float mag  = __expf(lamr * dt);
    Lbar[n]           = mag * __cosf(Lam_imag[n] * dt);
    Lbar[D_STATE + n] = mag * __sinf(Lam_imag[n] * dt);
  }
}

__global__ void k_split(const float* __restrict__ src,
                        bf16* __restrict__ hi, bf16* __restrict__ lo, size_t n) {
  size_t i = (size_t)blockIdx.x * blockDim.x + threadIdx.x;
  size_t stride = (size_t)gridDim.x * blockDim.x;
  for (; i < n; i += stride) {
    float v = src[i];
    bf16 h = f2bf(v);
    hi[i] = h;
    lo[i] = f2bf(v - (float)h);
  }
}

// Wb (128 x 1024): rows 0..63 = dt*B_real, rows 64..127 = dt*B_imag
__global__ void k_build_wb(const float* __restrict__ Br, const float* __restrict__ Bi,
                           const float* __restrict__ log_dt,
                           bf16* __restrict__ hi, bf16* __restrict__ lo) {
  int i = blockIdx.x * blockDim.x + threadIdx.x;  // 0..131071
  float dt = __expf(log_dt[0]);
  int n = i >> 10, h = i & 1023;
  float v = dt * (n < D_STATE ? Br[n * 1024 + h] : Bi[(n - D_STATE) * 1024 + h]);
  bf16 hv = f2bf(v);
  hi[i] = hv;
  lo[i] = f2bf(v - (float)hv);
}

// Wc (1024 x 128): cols 0..63 = C_real, cols 64..127 = -C_imag
__global__ void k_build_wc(const float* __restrict__ Cr, const float* __restrict__ Ci,
                           bf16* __restrict__ hi, bf16* __restrict__ lo) {
  int i = blockIdx.x * blockDim.x + threadIdx.x;  // 0..131071
  int h = i >> 7, n = i & 127;
  float v = (n < D_STATE) ? Cr[h * D_STATE + n] : -Ci[h * D_STATE + (n - D_STATE)];
  bf16 hv = f2bf(v);
  hi[i] = hv;
  lo[i] = f2bf(v - (float)hv);
}

// ---- sequential complex scan (512 independent chains) ----------------------
__global__ void k_scan(const float* __restrict__ BU, const float* __restrict__ Lbar,
                       bf16* __restrict__ Sh, bf16* __restrict__ Sl,
                       float* __restrict__ states) {
  int t = blockIdx.x * blockDim.x + threadIdx.x;   // 0..511
  int b = t >> 6, n = t & 63;
  float lr = Lbar[n], li = Lbar[D_STATE + n];
  float sr = 0.f, si = 0.f;
  const float* bu = BU + (size_t)b * SEQLEN * 128 + n;
  size_t orow = (size_t)b * SEQLEN * 128 + n;
  for (int l0 = 0; l0 < SEQLEN; l0 += 8) {
    float br[8], bi[8];
#pragma unroll
    for (int j = 0; j < 8; j++) {               // independent loads first
      br[j] = bu[(size_t)(l0 + j) * 128];
      bi[j] = bu[(size_t)(l0 + j) * 128 + 64];
    }
#pragma unroll
    for (int j = 0; j < 8; j++) {               // dependent FMA chain
      float nr = lr * sr - li * si + br[j];
      float ni = lr * si + li * sr + bi[j];
      sr = nr; si = ni;
      size_t r = orow + (size_t)(l0 + j) * 128;
      bf16 hr = f2bf(sr);
      Sh[r] = hr;           Sl[r] = f2bf(sr - (float)hr);
      bf16 hc = f2bf(si);
      Sh[r + 64] = hc;      Sl[r + 64] = f2bf(si - (float)hc);
    }
  }
  states[t] = sr;
  states[512 + t] = si;
}

// ---- generic NT GEMM, split-bf16, f32 accumulate ---------------------------
// C[m,n] = sum_k A[m,k]*W[n,k]; A/W as bf16 hi+lo pairs, row-major, ld=K.
// B tiles (64 x KSLAB, hi+lo) double-buffered in LDS via async global->LDS.
// MODE 0: Cf = acc   MODE 1: v = acc + aux[n]*Xf[m,n]; Ch/Cl = split(v)
// MODE 2: Cf = acc + aux[n]
template <int MODE>
__global__ __launch_bounds__(256)
void gemm_nt_split(const bf16* __restrict__ Ah, const bf16* __restrict__ Al,
                   const bf16* __restrict__ Bh, const bf16* __restrict__ Bl,
                   float* __restrict__ Cf, bf16* __restrict__ Ch, bf16* __restrict__ Cl,
                   const float* __restrict__ aux, const float* __restrict__ Xf,
                   int M, int N, int K) {
  __shared__ __align__(32) bf16 sB[2][2][64 * BPITCH];   // [buf][hi/lo] 36 KB

  const int tid  = threadIdx.x;
  const int lane = tid & 31;
  const int w    = tid >> 5;
  const int wm   = w & 3;          // 4 waves over M
  const int wn   = w >> 2;         // 2 waves over N
  const int m0   = blockIdx.y * 128 + wm * 32;
  const int nblk = blockIdx.x * 64;
  const int nw0  = wn * 32;        // wave's n offset inside block tile

  v8f acc[2][2] = {};
  const int arow0 = m0 + (lane & 15);
  const int S = K / KSLAB;

  stage_tile(Bh, &sB[0][0][0], nblk, K, 0, tid);
  stage_tile(Bl, &sB[0][1][0], nblk, K, 0, tid);

  for (int s = 0; s < S; ++s) {
    wait_async0();
    __syncthreads();                               // buf[s&1] ready for all waves
    if (s + 1 < S) {
      stage_tile(Bh, &sB[(s + 1) & 1][0][0], nblk, K, (s + 1) * KSLAB, tid);
      stage_tile(Bl, &sB[(s + 1) & 1][1][0], nblk, K, (s + 1) * KSLAB, tid);
      __builtin_prefetch(Ah + (size_t)arow0 * K + (s + 1) * KSLAB, 0, 1);
      __builtin_prefetch(Al + (size_t)arow0 * K + (s + 1) * KSLAB, 0, 1);
    }
    const bf16* tileH = &sB[s & 1][0][0];
    const bf16* tileL = &sB[s & 1][1][0];
    const int k0 = s * KSLAB;
#pragma unroll
    for (int kk = 0; kk < KSLAB; kk += 32) {
      bf16x16 a_h[2], a_l[2], b_h[2], b_l[2];
#pragma unroll
      for (int t = 0; t < 2; t++) {
        a_h[t] = load_a_frag(Ah, K, arow0 + 16 * t, k0 + kk, lane);
        a_l[t] = load_a_frag(Al, K, arow0 + 16 * t, k0 + kk, lane);
        b_h[t] = load_b_frag_lds(tileH, nw0 + 16 * t, kk, lane);
        b_l[t] = load_b_frag_lds(tileL, nw0 + 16 * t, kk, lane);
      }
#pragma unroll
      for (int tm = 0; tm < 2; tm++)
#pragma unroll
        for (int tn = 0; tn < 2; tn++) {
          v8f c = acc[tm][tn];
          // low-order terms first, hi*hi last (accuracy)
          c = __builtin_amdgcn_wmma_f32_16x16x32_bf16(false, a_l[tm], false, b_h[tn], (short)0, c, false, false);
          c = __builtin_amdgcn_wmma_f32_16x16x32_bf16(false, a_h[tm], false, b_l[tn], (short)0, c, false, false);
          c = __builtin_amdgcn_wmma_f32_16x16x32_bf16(false, a_h[tm], false, b_h[tn], (short)0, c, false, false);
          acc[tm][tn] = c;
        }
    }
    __syncthreads();                               // done reading buf[s&1]
  }

  // epilogue: lane l, vgpr j -> row = base + j + 8*(l/16), col = n + l%16
  const int crow = m0 + ((lane >> 4) << 3);
  const int ccol = lane & 15;
#pragma unroll
  for (int tm = 0; tm < 2; tm++)
#pragma unroll
    for (int tn = 0; tn < 2; tn++) {
      int colg = nblk + nw0 + tn * 16 + ccol;
#pragma unroll
      for (int j = 0; j < 8; j++) {
        int rowg = crow + tm * 16 + j;
        size_t off = (size_t)rowg * N + colg;
        float v = acc[tm][tn][j];
        if (MODE == 0) {
          Cf[off] = v;
        } else if (MODE == 1) {
          v += aux[colg] * Xf[off];
          bf16 h = f2bf(v);
          Ch[off] = h;
          Cl[off] = f2bf(v - (float)h);
        } else {
          Cf[off] = v + aux[colg];
        }
      }
    }
}

// ---------------------------------------------------------------------------
extern "C" void kernel_launch(void* const* d_in, const int* in_sizes, int n_in,
                              void* d_out, int out_size, void* d_ws, size_t ws_size,
                              hipStream_t stream) {
  const float* x       = (const float*)d_in[0];
  const float* logLr   = (const float*)d_in[1];
  const float* LamI    = (const float*)d_in[2];
  const float* B_real  = (const float*)d_in[3];
  const float* B_imag  = (const float*)d_in[4];
  const float* C_real  = (const float*)d_in[5];
  const float* C_imag  = (const float*)d_in[6];
  const float* Dvec    = (const float*)d_in[7];
  const float* log_dt  = (const float*)d_in[8];
  const float* W_out   = (const float*)d_in[9];
  const float* b_out   = (const float*)d_in[10];

  char* ws = (char*)d_ws;
  float* Lbar = (float*)(ws + O_LBAR);
  bf16 *WbH = (bf16*)(ws + O_WBH), *WbL = (bf16*)(ws + O_WBL);
  bf16 *WcH = (bf16*)(ws + O_WCH), *WcL = (bf16*)(ws + O_WCL);
  bf16 *WoH = (bf16*)(ws + O_WOH), *WoL = (bf16*)(ws + O_WOL);
  float* BU = (float*)(ws + O_BU);
  bf16 *Sh = (bf16*)(ws + O_SH), *Sl = (bf16*)(ws + O_SL);
  bf16 *XYh = (bf16*)(ws + O_XYH), *XYl = (bf16*)(ws + O_XYL);  // x split, reused as Y split
  float* out = (float*)d_out;
  float* states = out + OUT_ELEMS;   // final (sr,si): 2*512 floats

  // prep
  k_lambda<<<1, 64, 0, stream>>>(logLr, LamI, log_dt, Lbar);
  k_build_wb<<<512, 256, 0, stream>>>(B_real, B_imag, log_dt, WbH, WbL);
  k_build_wc<<<512, 256, 0, stream>>>(C_real, C_imag, WcH, WcL);
  k_split<<<2048, 256, 0, stream>>>(W_out, WoH, WoL, (size_t)1024 * 1024);
  k_split<<<8192, 256, 0, stream>>>(x, XYh, XYl, OUT_ELEMS);

  // GEMM1: BU(16384x128) = x(16384x1024) @ Wb^T
  gemm_nt_split<0><<<dim3(128 / 64, M_ROWS / 128), 256, 0, stream>>>(
      XYh, XYl, WbH, WbL, BU, nullptr, nullptr, nullptr, nullptr,
      M_ROWS, 128, 1024);

  // scan over L (512 chains), writes S hi/lo and final states
  k_scan<<<16, 32, 0, stream>>>(BU, Lbar, Sh, Sl, states);

  // GEMM2: Y(16384x1024) = S(16384x128) @ Wc^T + D*x   (Y overwrites x split)
  gemm_nt_split<1><<<dim3(1024 / 64, M_ROWS / 128), 256, 0, stream>>>(
      Sh, Sl, WcH, WcL, nullptr, XYh, XYl, Dvec, x,
      M_ROWS, 1024, 128);

  // GEMM3: OUT = Y @ W_out^T + b_out
  gemm_nt_split<2><<<dim3(1024 / 64, M_ROWS / 128), 256, 0, stream>>>(
      XYh, XYl, WoH, WoL, out, nullptr, nullptr, b_out, nullptr,
      M_ROWS, 1024, 1024);
}
